// TreeLSTMCell_52183852646691
// MI455X (gfx1250) — matrix operand order repaired and checked by reading, not compiled, not run on hardware
//
#include <hip/hip_runtime.h>
#include <hip/hip_bf16.h>

// ---------------------------------------------------------------------------
// TreeLSTM fused cell for MI455X (gfx1250, wave32, WMMA bf16 16x16x32).
//   f   = sigmoid(h_cat @ U_f_w[:512,:512] + U_f_b[:512])          [N,512]
//   c_r = sum_k f[:,k,:] * c_mail[:,k,:]                           [N,256]
//   iou = h_cat @ U_iou_w[:, :512].T + b_iou                       [N,768]
//   c   = sig(i)*tanh(u) + c_r ;  h_out = sig(o)*tanh(c)
// One WG = 256 threads = 8 waves, owns 128 rows. Wave w owns rows [16w,16w+16).
// For each of 16 h-tiles (16 output columns), each wave carries 5 accumulators
// (f0,f1,i,o,u) so the LSTM epilogue is fully fused in registers.
// ---------------------------------------------------------------------------

typedef __attribute__((ext_vector_type(16))) __bf16 v16bf;
typedef __attribute__((ext_vector_type(8)))  __bf16 v8bf;
typedef __attribute__((ext_vector_type(8)))  float  v8f;

#define WMMA_BF16(A, B, C) \
  __builtin_amdgcn_wmma_f32_16x16x32_bf16(false, (A), false, (B), (short)0, (C), false, false)

__device__ __forceinline__ __bf16 f2bf(float x) {
  union { float f; unsigned u; } v; v.f = x;
  unsigned r = (v.u + 0x7FFFu + ((v.u >> 16) & 1u)) >> 16;   // round-nearest-even
  union { unsigned short s; __bf16 b; } o; o.s = (unsigned short)r;
  return o.b;
}

__device__ __forceinline__ float sigmoidf_(float x) {
  return 1.0f / (1.0f + __expf(-x));
}
__device__ __forceinline__ float tanhf_(float x) {
  // tanh(x) = 1 - 2/(exp(2x)+1)
  return 1.0f - 2.0f / (__expf(2.0f * x) + 1.0f);
}

#define M_TILE   128
#define D_DIM    512          // K*H = 2*256 (reduction dim)
#define H_DIM    256
#define TEN_H    2560
#define A_STRIDE 520          // 512 + 8 bf16 pad (bank-conflict break)
#define B_ELEMS  (5 * 16 * 32 * 16)   // 5 regions * 16 k-chunks * 32 lanes * 16 elems

__global__ __launch_bounds__(256, 1)
void treelstm_fused_kernel(const float* __restrict__ h_mail,
                           const float* __restrict__ c_mail,
                           const float* __restrict__ Ufw,   // [2560,2560] row-major
                           const float* __restrict__ Ufb,   // [2560]
                           const float* __restrict__ Uiouw, // [768,2560] row-major
                           const float* __restrict__ biou,  // [768]
                           float* __restrict__ out_h,
                           float* __restrict__ out_c,
                           int nNodes) {
  __shared__ __align__(32) __bf16 A_lds[M_TILE * A_STRIDE];   // ~130 KB
  __shared__ __align__(32) __bf16 B_lds[B_ELEMS];             // 80 KB

  const int tid  = threadIdx.x;
  const int wave = tid >> 5;
  const int lane = tid & 31;
  const int n0   = blockIdx.x * M_TILE;

  // ---- Stage A: h_cat rows [n0, n0+128) -> bf16 LDS, row-major padded ----
  for (int idx = tid; idx < M_TILE * D_DIM; idx += 256) {
    int r = idx >> 9;           // row in tile
    int k = idx & 511;          // reduction index
    int n = n0 + r; if (n >= nNodes) n = nNodes - 1;   // clamp (stores guarded)
    A_lds[r * A_STRIDE + k] = f2bf(h_mail[(size_t)n * D_DIM + k]);
  }
  __syncthreads();

  const int myRow  = (lane & 15);            // M row within the wave's 16-row tile
  const int klo    = (lane < 16) ? 0 : 16;   // A-operand K sub-chunk select
  const __bf16* Abase = &A_lds[(wave * 16 + myRow) * A_STRIDE];

  for (int ht = 0; ht < 16; ++ht) {
    // ---- Stage B: 5 weight panels (f0,f1,i,o,u) for this h-tile, in this kernel's contiguous-k
    //      B-operand layout: [region][kchunk][lane(= khalf*16 + col)][elem(=k%16)]
    for (int idx = tid; idx < B_ELEMS; idx += 256) {
      int region = idx >> 13;                 // /8192
      int rem    = idx & 8191;
      int kc     = rem >> 9;                  // k-chunk 0..15
      int lane2  = (rem >> 4) & 31;
      int elem   = rem & 15;
      int khalf  = lane2 >> 4;
      int j      = lane2 & 15;                // column within 16-wide tile
      int k      = kc * 32 + khalf * 16 + elem;
      float w;
      if (region < 2) {
        // f GEMM uses un-transposed U_f_w slice: B[k][j] = Ufw[k, region*256 + ht*16 + j]
        int jj = region * 256 + ht * 16 + j;
        w = Ufw[(size_t)k * TEN_H + jj];
      } else {
        // iou GEMM uses transposed U_iou_w: B[k][j] = Uiouw[(region-2)*256 + ht*16 + j, k]
        int row = (region - 2) * 256 + ht * 16 + j;
        w = Uiouw[(size_t)row * TEN_H + k];
      }
      B_lds[idx] = f2bf(w);
    }
    __syncthreads();

    // ---- GEMM: 5 accumulators per wave, K = 512 in 16 steps of 32 ----
    v8f accF0 = {}, accF1 = {}, accI = {}, accO = {}, accU = {};
    for (int kc = 0; kc < 16; ++kc) {
      const int kb = kc * 32;
      v8bf alo = *(const v8bf*)(Abase + kb + klo);
      v8bf ahi = *(const v8bf*)(Abase + kb + klo + 8);
      v16bf a = __builtin_shufflevector(alo, ahi,
                  0,1,2,3,4,5,6,7,8,9,10,11,12,13,14,15);
      const __bf16* Bb = &B_lds[(kc * 32 + lane) * 16];
      v16bf b0 = *(const v16bf*)(Bb + 0 * 8192);
      v16bf b1 = *(const v16bf*)(Bb + 1 * 8192);
      v16bf b2 = *(const v16bf*)(Bb + 2 * 8192);
      v16bf b3 = *(const v16bf*)(Bb + 3 * 8192);
      v16bf b4 = *(const v16bf*)(Bb + 4 * 8192);
      accF0 = WMMA_BF16(a, b0, accF0);
      accF1 = WMMA_BF16(a, b1, accF1);
      accI  = WMMA_BF16(a, b2, accI);
      accO  = WMMA_BF16(a, b3, accO);
      accU  = WMMA_BF16(a, b4, accU);
    }

    // ---- Fused LSTM epilogue, entirely in registers ----
    // C/D tile layout: VGPR v, lanes 0-15 -> M=v; lanes 16-31 -> M=v+8; N=lane%16.
    const int h   = ht * 16 + (lane & 15);      // output feature index 0..255
    const float bf0 = Ufb[h];
    const float bf1 = Ufb[256 + h];
    const float bi  = biou[h];
    const float bo  = biou[256 + h];
    const float bu  = biou[512 + h];
    const int mHi   = (lane >= 16) ? 8 : 0;
#pragma unroll
    for (int v = 0; v < 8; ++v) {
      int n = n0 + wave * 16 + v + mHi;
      if (n < nNodes) {
        float cm0 = c_mail[(size_t)n * D_DIM + h];          // child 0
        float cm1 = c_mail[(size_t)n * D_DIM + H_DIM + h];  // child 1
        float f0  = sigmoidf_(accF0[v] + bf0);
        float f1  = sigmoidf_(accF1[v] + bf1);
        float cred = f0 * cm0 + f1 * cm1;
        float iv  = sigmoidf_(accI[v] + bi);
        float uv  = tanhf_(accU[v] + bu);
        float cc  = iv * uv + cred;
        float ho  = sigmoidf_(accO[v] + bo) * tanhf_(cc);
        out_h[(size_t)n * H_DIM + h] = ho;
        out_c[(size_t)n * H_DIM + h] = cc;
      }
    }
    __syncthreads();   // protect B_lds before next h-tile refill
  }
}

extern "C" void kernel_launch(void* const* d_in, const int* in_sizes, int n_in,
                              void* d_out, int out_size, void* d_ws, size_t ws_size,
                              hipStream_t stream) {
  const float* h_mail = (const float*)d_in[0];
  const float* c_mail = (const float*)d_in[1];
  const float* Ufw    = (const float*)d_in[2];
  const float* Ufb    = (const float*)d_in[3];
  const float* Uiouw  = (const float*)d_in[4];
  const float* biou   = (const float*)d_in[5];

  const int nNodes = in_sizes[0] / D_DIM;           // h_mail is [N, K, H] = [N, 512]
  float* out_h = (float*)d_out;                      // first output: h_out [N,256]
  float* out_c = out_h + (size_t)nNodes * H_DIM;     // second output: c [N,256]

  const int grid = (nNodes + M_TILE - 1) / M_TILE;   // 1563 WGs for N=200000
  treelstm_fused_kernel<<<grid, 256, 0, stream>>>(
      h_mail, c_mail, Ufw, Ufb, Uiouw, biou, out_h, out_c, nNodes);
}
